// GRUEncoder_66125316489315
// MI455X (gfx1250) — matrix operand order repaired and checked by reading, not compiled, not run on hardware
//
#include <hip/hip_runtime.h>
#include <hip/hip_bf16.h>

// ---------------------------------------------------------------------------
// GRU encoder for MI455X (gfx1250): persistent wave32 kernel, bf16 WMMA.
// B=128, K=512, I=512, H=512.
//  - Each wave owns one 16x16 (batch x hidden) tile; computes all 6 gate GEMM
//    tiles (r/z/n x {input, hidden}) so the GRU cell is wave-local and h stays
//    in fp32 registers for all 512 steps.
//  - The block's shared A-operands (X_t tile fp32->bf16, h tile bf16) are
//    staged in LDS once per step (8x traffic/convert reduction), h via
//    async global->LDS when the builtin exists.
//  - One device-wide barrier per step; h bf16 copy is ping-ponged.
// ---------------------------------------------------------------------------

#define B_DIM 128
#define K_STEPS 512
#define I_DIM 512
#define H_DIM 512
#define G3H (3 * H_DIM)

#define NBLOCKS 32
#define NTHREADS 256        // 8 waves/block; 32*8 = 256 waves = 8 x 32 tiles
#define LDS_STRIDE 520      // 512 + 8 bf16 pad -> conflict-free ds_load_b128

typedef __attribute__((ext_vector_type(16))) __bf16 bf16x16;
typedef __attribute__((ext_vector_type(8)))  __bf16 bf16x8;
typedef __attribute__((ext_vector_type(8)))  float  f32x8;
typedef __attribute__((ext_vector_type(4)))  float  f32x4;
typedef __attribute__((ext_vector_type(4)))  int    i32x4;

// Address-space-qualified int4 for the async global->LDS builtin
// (param 0 is int4* in addrspace(1) per the compiler diagnostic).
typedef __attribute__((address_space(1))) i32x4 as1_i32x4;
typedef __attribute__((address_space(3))) i32x4 as3_i32x4;

__device__ __forceinline__ bf16x16 frag_cat(bf16x8 lo, bf16x8 hi) {
  bf16x16 r;
#pragma unroll
  for (int i = 0; i < 8; ++i) { r[i] = lo[i]; r[i + 8] = hi[i]; }
  return r;
}

// A fragment (16x32 bf16). rowptr = &mat[row][0] (per-lane row = lane&15).
// Halves 0..7  <-> K = k0 + lh*8 + 0..7 ; halves 8..15 <-> K = k0+16+lh*8+0..7
__device__ __forceinline__ bf16x16 load_a_frag(const __bf16* rowptr, int lh, int k0) {
  bf16x8 lo = *(const bf16x8*)(rowptr + k0 + lh * 8);
  bf16x8 hi = *(const bf16x8*)(rowptr + k0 + 16 + lh * 8);
  return frag_cat(lo, hi);
}

// B fragment (32x16 bf16). colptr = &W[n_row][0] (W row == B column, lane&15).
// Halves 0..7  <-> K = k0 + lh*8 + 0..7 ; halves 8..15 <-> K = k0+16+lh*8+0..7
__device__ __forceinline__ bf16x16 load_b_frag(const __bf16* colptr, int lh, int k0) {
  bf16x8 lo = *(const bf16x8*)(colptr + k0 + lh * 8);
  bf16x8 hi = *(const bf16x8*)(colptr + k0 + 16 + lh * 8);
  return frag_cat(lo, hi);
}

__device__ __forceinline__ f32x8 wmma_bf16(bf16x16 a, bf16x16 b, f32x8 c) {
  return __builtin_amdgcn_wmma_f32_16x16x32_bf16(
      /*neg_a=*/false, a, /*neg_b=*/false, b,
      /*c_mod=*/(short)0, c, /*reuse_a=*/false, /*reuse_b=*/false);
}

__device__ __forceinline__ float sigmoidf_fast(float x) {
  return 1.0f / (1.0f + __expf(-x));
}

// Copy 64 bytes global->LDS for one thread (32 bf16), async when available.
__device__ __forceinline__ void stage_h64(const __bf16* src, __bf16* dst, bool* used_async) {
#if __has_builtin(__builtin_amdgcn_global_load_async_to_lds_b128)
#pragma unroll
  for (int i = 0; i < 4; ++i) {
    __builtin_amdgcn_global_load_async_to_lds_b128(
        (as1_i32x4*)(src + i * 8),
        (as3_i32x4*)(dst + i * 8),
        /*offset=*/0, /*cpol=*/0);
  }
  *used_async = true;
#else
#pragma unroll
  for (int i = 0; i < 4; ++i) {
    *(bf16x8*)(dst + i * 8) = *(const bf16x8*)(src + i * 8);
  }
  *used_async = false;
#endif
}

__device__ __forceinline__ void wait_async_done() {
#if __has_builtin(__builtin_amdgcn_global_load_async_to_lds_b128)
  asm volatile("s_wait_asynccnt 0x0" ::: "memory");
#endif
}

// Device-wide barrier (all NBLOCKS resident by construction).
__device__ __forceinline__ void grid_barrier(unsigned* cnt, unsigned* gen, unsigned nblocks) {
  __threadfence();  // publish this thread's global stores at device scope
  __syncthreads();
  if (threadIdx.x == 0) {
    unsigned g = __hip_atomic_load(gen, __ATOMIC_RELAXED, __HIP_MEMORY_SCOPE_AGENT);
    unsigned arrived =
        __hip_atomic_fetch_add(cnt, 1u, __ATOMIC_ACQ_REL, __HIP_MEMORY_SCOPE_AGENT);
    if (arrived == nblocks - 1) {
      __hip_atomic_store(cnt, 0u, __ATOMIC_RELAXED, __HIP_MEMORY_SCOPE_AGENT);
      __hip_atomic_store(gen, g + 1u, __ATOMIC_RELEASE, __HIP_MEMORY_SCOPE_AGENT);
    } else {
      while (__hip_atomic_load(gen, __ATOMIC_ACQUIRE, __HIP_MEMORY_SCOPE_AGENT) == g) {
        __builtin_amdgcn_s_sleep(2);
      }
    }
  }
  __syncthreads();
}

// ---------------------------------------------------------------------------
// Init: weights fp32 -> bf16, zero h ping-pong buffers, reset barrier state.
// ---------------------------------------------------------------------------
__global__ void gru_init_kernel(const float* __restrict__ wih, const float* __restrict__ whh,
                                __bf16* __restrict__ wih_bf, __bf16* __restrict__ whh_bf,
                                __bf16* __restrict__ hbuf, unsigned* __restrict__ bar,
                                int nw, int nh) {
  int i = blockIdx.x * blockDim.x + threadIdx.x;
  int stride = gridDim.x * blockDim.x;
  for (int k = i; k < nw; k += stride) {
    wih_bf[k] = (__bf16)wih[k];
    whh_bf[k] = (__bf16)whh[k];
  }
  for (int k = i; k < nh; k += stride) hbuf[k] = (__bf16)0.0f;
  if (i == 0) { bar[0] = 0u; bar[1] = 0u; }
}

// ---------------------------------------------------------------------------
// Persistent GRU.
// Block -> batch tile m = blockIdx>>2 (rows b0..b0+15), waves cover 8 hidden
// tiles j = (blockIdx&3)*8 + waveLocal. All 8 waves share the block's A tiles.
// ---------------------------------------------------------------------------
__global__ __launch_bounds__(NTHREADS, 1) void gru_persistent_kernel(
    const float* __restrict__ X,        // (B, K, I) fp32
    const float* __restrict__ b_ih,     // (3H)
    const float* __restrict__ b_hh,     // (3H)
    const __bf16* __restrict__ Wih_bf,  // (3H, I)
    const __bf16* __restrict__ Whh_bf,  // (3H, H)
    __bf16* __restrict__ hbuf,          // (2, B, H) ping-pong
    unsigned* __restrict__ bar,         // [cnt, gen]
    float* __restrict__ out,            // (B, H)
    unsigned nblocks) {
  __shared__ __bf16 lds_x[16 * LDS_STRIDE];  // X_t tile, bf16
  __shared__ __bf16 lds_h[16 * LDS_STRIDE];  // h tile, bf16

  const int lane = threadIdx.x & 31;
  const int l16  = lane & 15;
  const int lh   = lane >> 4;
  const int wloc = threadIdx.x >> 5;                     // 0..7
  const int b0   = (blockIdx.x >> 2) * 16;               // batch tile base
  const int j0   = (((blockIdx.x & 3) << 3) + wloc) * 16;  // hidden tile base

  // Per-lane constant pointers / scalars (weights are step-invariant).
  const __bf16* wih_col0; const __bf16* wih_col1; const __bf16* wih_col2;
  const __bf16* whh_col0; const __bf16* whh_col1; const __bf16* whh_col2;
  float bs0, bs1, bx2, bh2;  // pre-summed biases for r,z; separate for n
  {
    int r0 = 0 * H_DIM + j0 + l16;
    int r1 = 1 * H_DIM + j0 + l16;
    int r2 = 2 * H_DIM + j0 + l16;
    wih_col0 = Wih_bf + (size_t)r0 * I_DIM;  whh_col0 = Whh_bf + (size_t)r0 * H_DIM;
    wih_col1 = Wih_bf + (size_t)r1 * I_DIM;  whh_col1 = Whh_bf + (size_t)r1 * H_DIM;
    wih_col2 = Wih_bf + (size_t)r2 * I_DIM;  whh_col2 = Whh_bf + (size_t)r2 * H_DIM;
    bs0 = b_ih[r0] + b_hh[r0];
    bs1 = b_ih[r1] + b_hh[r1];
    bx2 = b_ih[r2];
    bh2 = b_hh[r2];
  }

  // Staging assignment: thread -> (row 0..15, 32-element segment 0..15).
  const int srow = threadIdx.x >> 4;
  const int sseg = threadIdx.x & 15;
  const float* xstage_base = X + ((size_t)(b0 + srow) * K_STEPS) * I_DIM + sseg * 32;
  __bf16* xstage_dst = &lds_x[srow * LDS_STRIDE + sseg * 32];
  __bf16* hstage_dst = &lds_h[srow * LDS_STRIDE + sseg * 32];
  const int hstage_off = (b0 + srow) * H_DIM + sseg * 32;

  // WMMA operand/result lane positions.
  const __bf16* lxrow = &lds_x[l16 * LDS_STRIDE];        // A rows from LDS
  const __bf16* lhrow = &lds_h[l16 * LDS_STRIDE];
  const int hpos_off = (b0 + 8 * lh) * H_DIM + j0 + l16; // C/D position (v=0)

  f32x8 hreg = {};  // this tile's h, fp32, register-resident across all steps

  for (int t = 0; t < K_STEPS; ++t) {
    const __bf16* hread  = hbuf + (size_t)(t & 1) * (B_DIM * H_DIM);
    __bf16*       hwrite = hbuf + (size_t)((t + 1) & 1) * (B_DIM * H_DIM);

    // ---- stage block-shared A tiles into LDS (once per block, not per wave)
    {
      const float* xs = xstage_base + (size_t)t * I_DIM;
#pragma unroll
      for (int c = 0; c < 32; c += 8) {
        f32x4 a = *(const f32x4*)(xs + c);
        f32x4 b = *(const f32x4*)(xs + c + 4);
        bf16x8 o;
#pragma unroll
        for (int i = 0; i < 4; ++i) { o[i] = (__bf16)a[i]; o[i + 4] = (__bf16)b[i]; }
        *(bf16x8*)(xstage_dst + c) = o;
      }
      bool used_async;
      stage_h64(hread + hstage_off, hstage_dst, &used_async);
      if (used_async) wait_async_done();
    }
    __syncthreads();

    // ---- 6 GEMM tiles: r/z/n gates x {input, hidden} projections
    f32x8 ax0 = {}, ax1 = {}, ax2 = {};
    f32x8 ah0 = {}, ah1 = {}, ah2 = {};
#pragma unroll 2
    for (int k0 = 0; k0 < I_DIM; k0 += 32) {
      bf16x16 aX = load_a_frag(lxrow, lh, k0);   // ds_load_b128 x2
      bf16x16 aH = load_a_frag(lhrow, lh, k0);
      ax0 = wmma_bf16(aX, load_b_frag(wih_col0, lh, k0), ax0);
      ah0 = wmma_bf16(aH, load_b_frag(whh_col0, lh, k0), ah0);
      ax1 = wmma_bf16(aX, load_b_frag(wih_col1, lh, k0), ax1);
      ah1 = wmma_bf16(aH, load_b_frag(whh_col1, lh, k0), ah1);
      ax2 = wmma_bf16(aX, load_b_frag(wih_col2, lh, k0), ax2);
      ah2 = wmma_bf16(aH, load_b_frag(whh_col2, lh, k0), ah2);
    }

    // ---- GRU cell, wave-local (z*h_prev hits this wave's own tile)
#pragma unroll
    for (int v = 0; v < 8; ++v) {
      float r = sigmoidf_fast(ax0[v] + ah0[v] + bs0);
      float z = sigmoidf_fast(ax1[v] + ah1[v] + bs1);
      float n = tanhf(ax2[v] + bx2 + r * (ah2[v] + bh2));
      float hn = (1.0f - z) * n + z * hreg[v];
      hreg[v] = hn;
      hwrite[hpos_off + v * H_DIM] = (__bf16)hn;  // bf16 copy for next step's A
    }

    grid_barrier(bar, bar + 1, nblocks);
  }

#pragma unroll
  for (int v = 0; v < 8; ++v) out[hpos_off + v * H_DIM] = hreg[v];
}

// ---------------------------------------------------------------------------
// Launch. Workspace layout (bytes, 256-aligned):
//   [0)        Wih_bf  3H*I*2 = 1,572,864
//   [1572864)  Whh_bf  3H*H*2 = 1,572,864
//   [3145728)  hbuf    2*B*H*2 =  262,144
//   [3407872)  barrier 2*u32
// Total ~3.4 MB.
// ---------------------------------------------------------------------------
extern "C" void kernel_launch(void* const* d_in, const int* in_sizes, int n_in,
                              void* d_out, int out_size, void* d_ws, size_t ws_size,
                              hipStream_t stream) {
  const float* X   = (const float*)d_in[0];
  const float* Wih = (const float*)d_in[1];
  const float* Whh = (const float*)d_in[2];
  const float* bih = (const float*)d_in[3];
  const float* bhh = (const float*)d_in[4];
  float* out = (float*)d_out;

  char* ws = (char*)d_ws;
  __bf16*   wih_bf = (__bf16*)(ws);
  __bf16*   whh_bf = (__bf16*)(ws + 1572864);
  __bf16*   hbuf   = (__bf16*)(ws + 3145728);
  unsigned* bar    = (unsigned*)(ws + 3407872);

  const int nw = G3H * I_DIM;        // 786432 weight elements per matrix
  const int nh = 2 * B_DIM * H_DIM;  // both h ping-pong buffers

  gru_init_kernel<<<512, 256, 0, stream>>>(Wih, Whh, wih_bf, whh_bf, hbuf, bar, nw, nh);
  gru_persistent_kernel<<<NBLOCKS, NTHREADS, 0, stream>>>(
      X, bih, bhh, wih_bf, whh_bf, hbuf, bar, out, (unsigned)NBLOCKS);
}
